// CrossBlock_27444841022016
// MI455X (gfx1250) — compile-verified, matrix operand order checked by reading, not run on hardware
//
#include <hip/hip_runtime.h>

// ---------------------------------------------------------------------------
// CrossBlock on gfx1250 (MI455X): all matmuls via v_wmma_f32_16x16x32_f16,
// LDS staging via GLOBAL_LOAD_ASYNC_TO_LDS_B128 (ASYNCcnt path).
// ---------------------------------------------------------------------------

typedef __attribute__((ext_vector_type(16))) _Float16 v16h;
typedef __attribute__((ext_vector_type(8)))  _Float16 v8h;
typedef __attribute__((ext_vector_type(4)))  _Float16 v4h;
typedef __attribute__((ext_vector_type(8)))  float    v8f;

#define DIM      768
#define NH       12
#define HD       64
#define MLP_DIM  3072
#define MQ       4096    // 4 * 32 * 32 query tokens
#define MKV      16384   // 4 * 64 * 64 kv tokens

// ---------------------------------------------------------------------------
// Async 16-byte global -> LDS copy (CDNA5 GLOBAL_LOAD_ASYNC_TO_LDS_B128).
// The low 32 bits of a generic pointer into __shared__ are the wave-relative
// LDS byte offset (aperture mapping, ISA 10.2), which is what the VDST
// operand of the async load expects. Tracked by ASYNCcnt.
// ---------------------------------------------------------------------------
__device__ __forceinline__ void async_copy16(const _Float16* gsrc, _Float16* ldst) {
    unsigned loff = (unsigned)(size_t)ldst;
    asm volatile("global_load_async_to_lds_b128 %0, %1, off"
                 :: "v"(loff), "v"(gsrc) : "memory");
}
__device__ __forceinline__ void wait_async0() {
    asm volatile("s_wait_asynccnt 0x0" ::: "memory");
}

// ---------------------------------------------------------------------------
// Fragment loader for 16-bit A/B operands of V_WMMA_F32_16X16X32_F16.
// ISA layout (CDNA5 ISA, 16-bit A-matrix 16x32):
//   lanes 0-15  : row = lane,    elems 0-7 -> k=0..7,  elems 8-15 -> k=16..23
//   lanes 16-31 : row = lane-16, elems 0-7 -> k=8..15, elems 8-15 -> k=24..31
// B (32x16) is symmetric with "row" = output column, so B is sourced from a
// K-major (transposed) tile with the same loader. Two ds_load_b128 per frag.
// ---------------------------------------------------------------------------
__device__ __forceinline__ v16h ld_frag(const _Float16* base, int ld) {
    const int lane = threadIdx.x & 31;
    const int idx  = lane & 15;
    const int koff = (lane >> 4) << 3;   // 0 or 8
    const _Float16* p = base + idx * ld + koff;
    v8h lo = *(const v8h*)(p);
    v8h hi = *(const v8h*)(p + 16);
    return __builtin_shufflevector(lo, hi, 0,1,2,3,4,5,6,7,8,9,10,11,12,13,14,15);
}

__device__ __forceinline__ v8f wmma_f16(v16h a, v16h b, v8f c) {
    return __builtin_amdgcn_wmma_f32_16x16x32_f16(false, a, false, b,
                                                  (short)0, c, false, false);
}

// ---------------------------------------------------------------------------
// Prep kernels: f32->f16 cast, and f32 [K][N] -> f16 [N][K] transpose-cast.
// ---------------------------------------------------------------------------
__global__ __launch_bounds__(256) void cb_cast_kernel(const float* __restrict__ in,
                                                      _Float16* __restrict__ out, int n4) {
    int i = blockIdx.x * 256 + threadIdx.x;
    if (i < n4) {
        float4 v = ((const float4*)in)[i];
        v4h o = { (_Float16)v.x, (_Float16)v.y, (_Float16)v.z, (_Float16)v.w };
        ((v4h*)out)[i] = o;
    }
}

__global__ __launch_bounds__(256) void cb_tcast_kernel(const float* __restrict__ in,
                                                       _Float16* __restrict__ out,
                                                       int K, int N) {
    long long i = (long long)blockIdx.x * 256 + threadIdx.x;
    if (i < (long long)K * N) {
        int k = (int)(i / N), n = (int)(i % N);
        out[(size_t)n * K + k] = (_Float16)in[i];
    }
}

// ---------------------------------------------------------------------------
// LayerNorm over 768 columns, one row per 256-thread block, f16 output.
// ---------------------------------------------------------------------------
__global__ __launch_bounds__(256) void cb_ln_kernel(const float* __restrict__ x,
                                                    const float* __restrict__ w,
                                                    const float* __restrict__ b,
                                                    _Float16* __restrict__ y) {
    __shared__ float red[256];
    __shared__ float red2[256];
    const int row = blockIdx.x, tid = threadIdx.x;
    const float* xr = x + (size_t)row * DIM;
    float v[3], s = 0.f, ss = 0.f;
    #pragma unroll
    for (int i = 0; i < 3; ++i) { v[i] = xr[tid + i * 256]; s += v[i]; ss += v[i] * v[i]; }
    red[tid] = s; red2[tid] = ss;
    __syncthreads();
    for (int o = 128; o > 0; o >>= 1) {
        if (tid < o) { red[tid] += red[tid + o]; red2[tid] += red2[tid + o]; }
        __syncthreads();
    }
    const float mu  = red[0] * (1.f / DIM);
    const float var = red2[0] * (1.f / DIM) - mu * mu;
    const float rs  = rsqrtf(var + 1e-5f);
    #pragma unroll
    for (int i = 0; i < 3; ++i) {
        int c = tid + i * 256;
        y[(size_t)row * DIM + c] = (_Float16)((v[i] - mu) * rs * w[c] + b[c]);
    }
}

// ---------------------------------------------------------------------------
// Tiled WMMA GEMM:  C[M,N] = A[M,K] @ Bt[N,K]^T  (+bias, epilogue variants)
//   block tile 128x128, 8 waves (4 row x 2 col), wave tile 32x64 (2x4 frags),
//   KT=32, async global->LDS staging. 8 WMMAs per wave per k-step against
//   12 ds_load_b128 (vs 4:8 for a 32x32 wave tile).
// mode 0: outh = f16((acc + bias) * scale)
// mode 1: outf = acc + bias + res          (f32 residual store)
// mode 2: outh = f16(gelu_exact(acc+bias)) (erf-based GELU)
// ---------------------------------------------------------------------------
#define BM  128
#define BN  128
#define KT  32
#define LDT 40   // padded LDS leading dim (halfs); 80B rows keep 16B alignment

__global__ __launch_bounds__(256) void cb_gemm_kernel(
    const _Float16* __restrict__ A, const _Float16* __restrict__ Bt,
    const float* __restrict__ bias, const float* __restrict__ res,
    _Float16* __restrict__ outh, float* __restrict__ outf,
    int M, int N, int K, float scale, int mode)
{
    __shared__ __align__(16) _Float16 As[BM * LDT];
    __shared__ __align__(16) _Float16 Bs[BN * LDT];

    const int tid  = threadIdx.x;
    const int wave = tid >> 5;
    const int wm   = wave & 3;        // 0..3 -> 32-row slice
    const int wn   = wave >> 2;       // 0..1 -> 64-col slice
    const int m0   = blockIdx.y * BM;
    const int n0   = blockIdx.x * BN;
    const int lane = tid & 31, idx = lane & 15, rhi = (lane >> 4) << 3;

    const int ar  = tid >> 2;            // 0..63
    const int akc = (tid & 3) << 3;      // 0,8,16,24

    v8f acc[2][4] = {};

    for (int k0 = 0; k0 < K; k0 += KT) {
        __syncthreads();   // WAR: previous iteration's fragment reads done
        async_copy16(&A [(size_t)(m0 + ar)      * K + k0 + akc], &As[ ar       * LDT + akc]);
        async_copy16(&A [(size_t)(m0 + ar + 64) * K + k0 + akc], &As[(ar + 64) * LDT + akc]);
        async_copy16(&Bt[(size_t)(n0 + ar)      * K + k0 + akc], &Bs[ ar       * LDT + akc]);
        async_copy16(&Bt[(size_t)(n0 + ar + 64) * K + k0 + akc], &Bs[(ar + 64) * LDT + akc]);
        if (k0 + KT < K)  // warm caches for the next A tile (global_prefetch_b8)
            __builtin_prefetch(&A[(size_t)(m0 + ar) * K + k0 + KT + akc], 0, 0);
        wait_async0();
        __syncthreads();

        v16h a0 = ld_frag(&As[(wm * 32 +  0) * LDT], LDT);
        v16h a1 = ld_frag(&As[(wm * 32 + 16) * LDT], LDT);
        #pragma unroll
        for (int tc = 0; tc < 4; ++tc) {
            v16h bf = ld_frag(&Bs[(wn * 64 + tc * 16) * LDT], LDT);
            acc[0][tc] = wmma_f16(a0, bf, acc[0][tc]);
            acc[1][tc] = wmma_f16(a1, bf, acc[1][tc]);
        }
    }

    #pragma unroll
    for (int tr = 0; tr < 2; ++tr)
    #pragma unroll
    for (int tc = 0; tc < 4; ++tc)
    #pragma unroll
    for (int r = 0; r < 8; ++r) {
        const int gm = m0 + wm * 32 + tr * 16 + rhi + r;
        const int gn = n0 + wn * 64 + tc * 16 + idx;
        float val = acc[tr][tc][r] + bias[gn];
        if (mode == 0) {
            outh[(size_t)gm * N + gn] = (_Float16)(val * scale);
        } else if (mode == 1) {
            outf[(size_t)gm * N + gn] = val + res[(size_t)gm * N + gn];
        } else {
            float g = 0.5f * val * (1.0f + erff(val * 0.70710678118654752f));
            outh[(size_t)gm * N + gn] = (_Float16)g;
        }
    }
}

// ---------------------------------------------------------------------------
// Flash-attention per (batch, head, 64-query tile). 4 waves (2x2).
//   q: [MQ][DIM] f16 (already scaled by HD^-0.5 in q GEMM epilogue)
//   kv: [MKV][2*DIM] f16 (k at col h*64, v at col DIM + h*64)
//   o: [MQ][DIM] f16
// Online softmax: 64 threads own one q row each (running max/sum in regs),
// P staged to LDS f16 for the A operand of P@V, V staged transposed for B.
// Q/K tiles staged with async-to-LDS loads.
// ---------------------------------------------------------------------------
#define LDV   72   // padded f16 LDS leading dim (144B rows, 16B aligned)
#define LDS_S 68   // padded f32 leading dim for scores

__global__ __launch_bounds__(128) void cb_attn_kernel(
    const _Float16* __restrict__ qh, const _Float16* __restrict__ kvh,
    _Float16* __restrict__ oh)
{
    __shared__ __align__(16) _Float16 Qs [64 * LDV];
    __shared__ __align__(16) _Float16 Ks [64 * LDV];
    __shared__ __align__(16) _Float16 Vts[64 * LDV];
    __shared__ __align__(16) _Float16 Ps [64 * LDV];
    __shared__ float Ss[64 * LDS_S];
    __shared__ float rowscale[64];
    __shared__ float rowsum[64];

    const int tid  = threadIdx.x;
    const int qt = blockIdx.x, h = blockIdx.y, b = blockIdx.z;
    const int wave = tid >> 5;
    const int wm = wave >> 1, wn = wave & 1;         // 2x2 wave grid
    const int lane = tid & 31, idx = lane & 15, rhi = (lane >> 4) << 3;

    const size_t qrow0 = (size_t)b * 1024 + (size_t)qt * 64;
    const size_t krow0 = (size_t)b * 4096;

    // stage Q tile (64 x 64 halfs) asynchronously; first in-loop wait covers it
    #pragma unroll
    for (int it = 0; it < 4; ++it) {
        int c = tid + it * 128;
        int r = c >> 3, ch = (c & 7) << 3;
        async_copy16(&qh[(qrow0 + r) * DIM + h * HD + ch], &Qs[r * LDV + ch]);
    }

    float m_reg = -1e30f, l_reg = 0.f;
    v8f oacc[2][2] = {};

    for (int j = 0; j < 64; ++j) {               // 64 kv chunks of 64 tokens
        __syncthreads();
        #pragma unroll
        for (int it = 0; it < 4; ++it) {         // K chunk, token-major, async
            int c = tid + it * 128;
            int r = c >> 3, ch = (c & 7) << 3;
            async_copy16(&kvh[(krow0 + j * 64 + r) * (2 * DIM) + h * HD + ch],
                         &Ks[r * LDV + ch]);
        }
        for (int e = tid; e < 64 * 64; e += 128) {  // V chunk, transposed [d][tok]
            int tok = e >> 6, d = e & 63;
            Vts[d * LDV + tok] =
                kvh[(krow0 + j * 64 + tok) * (2 * DIM) + DIM + h * HD + d];
        }
        wait_async0();
        __syncthreads();

        // S = Q @ K^T   (k-dim = HD = 64 -> two WMMA k-steps)
        v8f sacc[2][2] = {};
        #pragma unroll
        for (int kk = 0; kk < 64; kk += 32) {
            v16h a0 = ld_frag(&Qs[(wm * 32 +  0) * LDV + kk], LDV);
            v16h a1 = ld_frag(&Qs[(wm * 32 + 16) * LDV + kk], LDV);
            v16h b0 = ld_frag(&Ks[(wn * 32 +  0) * LDV + kk], LDV);
            v16h b1 = ld_frag(&Ks[(wn * 32 + 16) * LDV + kk], LDV);
            sacc[0][0] = wmma_f16(a0, b0, sacc[0][0]);
            sacc[0][1] = wmma_f16(a0, b1, sacc[0][1]);
            sacc[1][0] = wmma_f16(a1, b0, sacc[1][0]);
            sacc[1][1] = wmma_f16(a1, b1, sacc[1][1]);
        }
        #pragma unroll
        for (int tr = 0; tr < 2; ++tr)
        #pragma unroll
        for (int tc = 0; tc < 2; ++tc)
        #pragma unroll
        for (int r = 0; r < 8; ++r) {
            int row = wm * 32 + tr * 16 + rhi + r;
            int col = wn * 32 + tc * 16 + idx;
            Ss[row * LDS_S + col] = sacc[tr][tc][r];
        }
        __syncthreads();

        // online softmax: thread t owns q row t
        if (tid < 64) {
            float cmax = -1e30f;
            for (int c = 0; c < 64; ++c) cmax = fmaxf(cmax, Ss[tid * LDS_S + c]);
            float mnew = fmaxf(m_reg, cmax);
            float sc   = __expf(m_reg - mnew);
            float ps = 0.f;
            for (int c = 0; c < 64; ++c) {
                float p = __expf(Ss[tid * LDS_S + c] - mnew);
                Ps[tid * LDV + c] = (_Float16)p;
                ps += p;
            }
            l_reg = l_reg * sc + ps;
            m_reg = mnew;
            rowscale[tid] = sc;
            rowsum[tid]   = l_reg;
        }
        __syncthreads();

        // rescale running O, then O += P @ V
        #pragma unroll
        for (int tr = 0; tr < 2; ++tr)
        #pragma unroll
        for (int r = 0; r < 8; ++r) {
            float sc = rowscale[wm * 32 + tr * 16 + rhi + r];
            oacc[tr][0][r] *= sc;
            oacc[tr][1][r] *= sc;
        }
        #pragma unroll
        for (int kk = 0; kk < 64; kk += 32) {
            v16h a0 = ld_frag(&Ps[(wm * 32 +  0) * LDV + kk], LDV);
            v16h a1 = ld_frag(&Ps[(wm * 32 + 16) * LDV + kk], LDV);
            v16h b0 = ld_frag(&Vts[(wn * 32 +  0) * LDV + kk], LDV);
            v16h b1 = ld_frag(&Vts[(wn * 32 + 16) * LDV + kk], LDV);
            oacc[0][0] = wmma_f16(a0, b0, oacc[0][0]);
            oacc[0][1] = wmma_f16(a0, b1, oacc[0][1]);
            oacc[1][0] = wmma_f16(a1, b0, oacc[1][0]);
            oacc[1][1] = wmma_f16(a1, b1, oacc[1][1]);
        }
    }
    __syncthreads();

    #pragma unroll
    for (int tr = 0; tr < 2; ++tr)
    #pragma unroll
    for (int r = 0; r < 8; ++r) {
        const int row = wm * 32 + tr * 16 + rhi + r;
        const float inv = 1.f / rowsum[row];
        #pragma unroll
        for (int tc = 0; tc < 2; ++tc) {
            const int d = wn * 32 + tc * 16 + idx;
            oh[(qrow0 + row) * DIM + h * HD + d] = (_Float16)(oacc[tr][tc][r] * inv);
        }
    }
}

// ---------------------------------------------------------------------------
// Host-side orchestration (graph-capture safe: stream launches only).
// ---------------------------------------------------------------------------
extern "C" void kernel_launch(void* const* d_in, const int* in_sizes, int n_in,
                              void* d_out, int out_size, void* d_ws, size_t ws_size,
                              hipStream_t stream) {
    const float* xq     = (const float*)d_in[0];
    const float* xkv    = (const float*)d_in[1];
    const float* n1w    = (const float*)d_in[2];
    const float* n1b    = (const float*)d_in[3];
    const float* kv_w   = (const float*)d_in[4];
    const float* kv_b   = (const float*)d_in[5];
    const float* q_w    = (const float*)d_in[6];
    const float* q_b    = (const float*)d_in[7];
    const float* proj_w = (const float*)d_in[8];
    const float* proj_b = (const float*)d_in[9];
    const float* n2w    = (const float*)d_in[10];
    const float* n2b    = (const float*)d_in[11];
    const float* w1     = (const float*)d_in[12];
    const float* b1     = (const float*)d_in[13];
    const float* w2     = (const float*)d_in[14];
    const float* b2     = (const float*)d_in[15];

    char* ws = (char*)d_ws;
    size_t off = 0;
    auto carve = [&](size_t bytes) -> char* {
        char* p = ws + off;
        off += (bytes + 255) & ~(size_t)255;
        return p;
    };

    _Float16* xkv_h  = (_Float16*)carve((size_t)MKV * DIM * 2);       // reused as mlp hidden
    _Float16* kv_h   = (_Float16*)carve((size_t)MKV * 2 * DIM * 2);
    _Float16* xn_h   = (_Float16*)carve((size_t)MQ * DIM * 2);        // ln1 then ln2 output
    _Float16* q_h    = (_Float16*)carve((size_t)MQ * DIM * 2);
    _Float16* o_h    = (_Float16*)carve((size_t)MQ * DIM * 2);
    float*    x_f    = (float*)   carve((size_t)MQ * DIM * 4);
    _Float16* kv_wt  = (_Float16*)carve((size_t)(2 * DIM) * DIM * 2);
    _Float16* q_wt   = (_Float16*)carve((size_t)DIM * DIM * 2);
    _Float16* proj_wt= (_Float16*)carve((size_t)DIM * DIM * 2);
    _Float16* w1t    = (_Float16*)carve((size_t)MLP_DIM * DIM * 2);
    _Float16* w2t    = (_Float16*)carve((size_t)DIM * MLP_DIM * 2);
    _Float16* h_h    = xkv_h;   // alias: xkv_h dead after kv GEMM, same size

    // --- prep: casts / weight transposes -----------------------------------
    cb_cast_kernel<<<(MKV * DIM / 4 + 255) / 256, 256, 0, stream>>>(xkv, xkv_h, MKV * DIM / 4);
    cb_tcast_kernel<<<(DIM * 2 * DIM + 255) / 256, 256, 0, stream>>>(kv_w, kv_wt, DIM, 2 * DIM);
    cb_tcast_kernel<<<(DIM * DIM + 255) / 256, 256, 0, stream>>>(q_w, q_wt, DIM, DIM);
    cb_tcast_kernel<<<(DIM * DIM + 255) / 256, 256, 0, stream>>>(proj_w, proj_wt, DIM, DIM);
    cb_tcast_kernel<<<(DIM * MLP_DIM + 255) / 256, 256, 0, stream>>>(w1, w1t, DIM, MLP_DIM);
    cb_tcast_kernel<<<(MLP_DIM * DIM + 255) / 256, 256, 0, stream>>>(w2, w2t, MLP_DIM, DIM);

    // --- LN1 ---------------------------------------------------------------
    cb_ln_kernel<<<MQ, 256, 0, stream>>>(xq, n1w, n1b, xn_h);

    // --- kv = xkv @ kv_w + kv_b  (f16 out, [MKV][1536]) --------------------
    cb_gemm_kernel<<<dim3(2 * DIM / BN, MKV / BM), 256, 0, stream>>>(
        xkv_h, kv_wt, kv_b, nullptr, kv_h, nullptr, MKV, 2 * DIM, DIM, 1.0f, 0);

    // --- q = (ln1 @ q_w + q_b) * HD^-0.5  (f16 out) ------------------------
    cb_gemm_kernel<<<dim3(DIM / BN, MQ / BM), 256, 0, stream>>>(
        xn_h, q_wt, q_b, nullptr, q_h, nullptr, MQ, DIM, DIM, 0.125f, 0);

    // --- flash attention ---------------------------------------------------
    cb_attn_kernel<<<dim3(16, NH, 4), 128, 0, stream>>>(q_h, kv_h, o_h);

    // --- x = xq + (o @ proj_w + proj_b)  (f32 out) -------------------------
    cb_gemm_kernel<<<dim3(DIM / BN, MQ / BM), 256, 0, stream>>>(
        o_h, proj_wt, proj_b, xq, nullptr, x_f, MQ, DIM, DIM, 1.0f, 1);

    // --- LN2 ---------------------------------------------------------------
    cb_ln_kernel<<<MQ, 256, 0, stream>>>(x_f, n2w, n2b, xn_h);

    // --- h = gelu(ln2 @ w1 + b1)  (f16 out) --------------------------------
    cb_gemm_kernel<<<dim3(MLP_DIM / BN, MQ / BM), 256, 0, stream>>>(
        xn_h, w1t, b1, nullptr, h_h, nullptr, MQ, MLP_DIM, DIM, 1.0f, 2);

    // --- out = x + (h @ w2 + b2)  (f32 out, final) -------------------------
    cb_gemm_kernel<<<dim3(DIM / BN, MQ / BM), 256, 0, stream>>>(
        h_h, w2t, b2, x_f, nullptr, (float*)d_out, MQ, DIM, MLP_DIM, 1.0f, 1);

    (void)in_sizes; (void)n_in; (void)out_size; (void)ws_size;
}
